// AttentionBlock_47047071760929
// MI455X (gfx1250) — matrix operand order repaired and checked by reading, not compiled, not run on hardware
//
#include <hip/hip_runtime.h>
#include <math.h>

// ---------------------------------------------------------------------------
// AttentionBlock for MI455X (gfx1250, wave32, WMMA 16x16x32 bf16)
//
// B=8, T=2048, D=512, K=V=512.  Softmax is over the QUERY axis (axis=1) per
// the reference, so normalization is per-column j over rows i>=j.  Plan:
//   1) split-convert x and weights to bf16 hi/lo pairs (near-fp32 accuracy via
//      3-term WMMA: hh + hl + lh, fp32 accumulate)
//   2) projections Q,K (row-major bf16 split) and V (transposed [b][v][t]
//      bf16 split) with fused bias
//   3) logits = Q.K^T, fp32, lower-triangular blocks only (134MB, fits in the
//      192MB L2 -> written once, read twice, cheap)
//   4) column stats: online max + sum(exp) over i>=j per column j, with
//      global_prefetch_b8 ahead of the serial scan
//   5) P-build: P = exp(l-m_j)*inv(s_j*sqrt(512)) computed ONCE per logit,
//      stored as bf16 hi/lo (masked region zeroed)
//   6) PV + residual: plain split-bf16 GEMM P @ V^T + x, 128 v-cols per wave
// ---------------------------------------------------------------------------

#define BB 8
#define TT 2048
#define DD 512
#define NROW (BB * TT)

typedef __attribute__((ext_vector_type(16))) __bf16 v16bf;
typedef __attribute__((ext_vector_type(8)))  __bf16 v8bf;
typedef __attribute__((ext_vector_type(8)))  float  v8f;

__device__ __forceinline__ v16bf cat16(v8bf lo, v8bf hi) {
  return __builtin_shufflevector(lo, hi, 0, 1, 2, 3, 4, 5, 6, 7,
                                         8, 9, 10, 11, 12, 13, 14, 15);
}

// A fragment: 16x32 bf16, rows row0..row0+15, K-slice k0..k0+31.
// lane&15 = M row; lanes 0-15 hold K {k0..k0+7, k0+16..k0+23},
// lanes 16-31 hold K {k0+8..k0+15, k0+24..k0+31}.  Two 16B loads.
__device__ __forceinline__ v16bf load_frag_a(const __bf16* base, int ld,
                                             int row0, int k0, int lane) {
  int r = lane & 15;
  int h = (lane >> 4) << 3;  // 0 or 8
  const __bf16* p = base + (size_t)(row0 + r) * ld + (k0 + h);
  v8bf lo = *(const v8bf*)p;
  v8bf hi = *(const v8bf*)(p + 16);
  return cat16(lo, hi);
}

// B fragment: 32x16 bf16 (K x N).  lane&15 = N column, mapped to memory row
// (n0 + n) of a row-major-over-K operand; lanes 0-15 hold K {k0..k0+7,
// k0+16..k0+23}, lanes 16-31 hold K {k0+8..k0+15, k0+24..k0+31}.  Two 16B loads.
__device__ __forceinline__ v16bf load_frag_b(const __bf16* base, int ld,
                                             int n0, int k0, int lane) {
  int n = lane & 15;
  int ks = k0 + ((lane >> 4) << 3);
  const __bf16* p = base + (size_t)(n0 + n) * ld + ks;
  v8bf lo = *(const v8bf*)p;
  v8bf hi = *(const v8bf*)(p + 16);
  return cat16(lo, hi);
}

#define WMMA_BF16(A, Bm, C)                                                   \
  __builtin_amdgcn_wmma_f32_16x16x32_bf16(false, (A), false, (Bm), (short)0,  \
                                          (C), false, false)

// --------------------------- kernel 1: split convert -----------------------
__global__ __launch_bounds__(256) void cvt_split_kernel(
    const float* __restrict__ in, __bf16* __restrict__ oh,
    __bf16* __restrict__ ol, int n) {
  int i = blockIdx.x * blockDim.x + threadIdx.x;
  if (i < n) {
    float f = in[i];
    __bf16 h = (__bf16)f;
    oh[i] = h;
    ol[i] = (__bf16)(f - (float)h);
  }
}

// --------------------------- kernel 2: projection --------------------------
// Y = X W^T + b.  Block = 4 waves; wave w owns rows m0 = bx*64 + w*16 and all
// 64 columns n0 = by*64 .. +63 (4 accumulator tiles).  Output is written as a
// bf16 hi/lo split; trans=1 writes V transposed to [b][n][t].
__global__ __launch_bounds__(128) void proj_wmma_kernel(
    const __bf16* __restrict__ xh, const __bf16* __restrict__ xl,
    const __bf16* __restrict__ wh, const __bf16* __restrict__ wl,
    const float* __restrict__ bias, __bf16* __restrict__ oh,
    __bf16* __restrict__ ol, int trans) {
  int lane = threadIdx.x & 31;
  int wv = threadIdx.x >> 5;
  int m0 = blockIdx.x * 64 + wv * 16;
  int n0 = blockIdx.y * 64;

  v8f acc[4] = {};
  for (int k0 = 0; k0 < DD; k0 += 32) {
    v16bf ah = load_frag_a(xh, DD, m0, k0, lane);
    v16bf al = load_frag_a(xl, DD, m0, k0, lane);
#pragma unroll
    for (int t = 0; t < 4; ++t) {
      v16bf bh = load_frag_b(wh, DD, n0 + t * 16, k0, lane);
      v16bf bl = load_frag_b(wl, DD, n0 + t * 16, k0, lane);
      acc[t] = WMMA_BF16(ah, bh, acc[t]);
      acc[t] = WMMA_BF16(ah, bl, acc[t]);
      acc[t] = WMMA_BF16(al, bh, acc[t]);
    }
  }

  int rbase = (lane >> 4) * 8;
#pragma unroll
  for (int t = 0; t < 4; ++t) {
    int c = n0 + t * 16 + (lane & 15);
    float bvv = bias[c];
#pragma unroll
    for (int r = 0; r < 8; ++r) {
      int m = m0 + rbase + r;  // global row (b*T + t_idx)
      float y = acc[t][r] + bvv;
      __bf16 h = (__bf16)y;
      __bf16 l = (__bf16)(y - (float)h);
      size_t idx;
      if (trans)
        idx = ((size_t)(m >> 11) * DD + c) * TT + (m & (TT - 1));
      else
        idx = (size_t)m * DD + c;
      oh[idx] = h;
      ol[idx] = l;
    }
  }
}

// --------------------------- kernel 3: logits ------------------------------
// logits[b,i,j] = q_i . k_j for block-lower-triangular tiles (bj <= bi).
__global__ __launch_bounds__(128) void logits_wmma_kernel(
    const __bf16* __restrict__ qh, const __bf16* __restrict__ ql,
    const __bf16* __restrict__ kh, const __bf16* __restrict__ kl,
    float* __restrict__ logits) {
  int bi = blockIdx.x, bj = blockIdx.y, b = blockIdx.z;
  if (bj > bi) return;
  int lane = threadIdx.x & 31;
  int wv = threadIdx.x >> 5;
  int i0 = bi * 64 + wv * 16;
  int j0 = bj * 64;
  int rowQ = b * TT + i0;
  int rowK = b * TT + j0;

  v8f acc[4] = {};
  for (int k0 = 0; k0 < DD; k0 += 32) {
    v16bf ah = load_frag_a(qh, DD, rowQ, k0, lane);
    v16bf al = load_frag_a(ql, DD, rowQ, k0, lane);
#pragma unroll
    for (int t = 0; t < 4; ++t) {
      v16bf bh = load_frag_b(kh, DD, rowK + t * 16, k0, lane);
      v16bf bl = load_frag_b(kl, DD, rowK + t * 16, k0, lane);
      acc[t] = WMMA_BF16(ah, bh, acc[t]);
      acc[t] = WMMA_BF16(ah, bl, acc[t]);
      acc[t] = WMMA_BF16(al, bh, acc[t]);
    }
  }

  float* lbase = logits + (size_t)b * TT * TT;
  int rbase = (lane >> 4) * 8;
#pragma unroll
  for (int t = 0; t < 4; ++t) {
    int j = j0 + t * 16 + (lane & 15);
#pragma unroll
    for (int r = 0; r < 8; ++r) {
      int i = i0 + rbase + r;
      lbase[(size_t)i * TT + j] = acc[t][r];
    }
  }
}

// --------------------------- kernel 4: column stats ------------------------
// Softmax is over axis=1 (queries): per column j, over rows i >= j.
// Online max/sum in one pass; reads are coalesced along j; prefetch 16 rows
// ahead (global_prefetch_b8) to cover L2 latency on the serial scan.
__global__ __launch_bounds__(256) void colstats_kernel(
    const float* __restrict__ logits, float* __restrict__ colmax,
    float* __restrict__ colscale) {
  int b = blockIdx.y;
  int j = blockIdx.x * blockDim.x + threadIdx.x;
  const float* lb = logits + (size_t)b * TT * TT;
  float m = -__builtin_inff();
  float s = 0.f;
  for (int i = j; i < TT; ++i) {
    if (i + 16 < TT) __builtin_prefetch(&lb[(size_t)(i + 16) * TT + j], 0, 1);
    float v = lb[(size_t)i * TT + j];
    if (v > m) {
      s = s * expf(m - v) + 1.f;
      m = v;
    } else {
      s += expf(v - m);
    }
  }
  colmax[b * TT + j] = m;
  // fold the post-softmax 1/sqrt(K_SIZE) scale into the inverse sum
  colscale[b * TT + j] = 1.f / (s * 22.62741699796952f);
}

// --------------------------- kernel 5: P build -----------------------------
// P[b,i,j] = (j<=i) ? exp(l-m_j)*scale_j : 0, split to bf16 hi/lo.  One block
// per (i,b); thread tid covers j0 = tid*8 (one b128 store per array).  Only
// j < roundup32(i+16) is ever read by the PV GEMM, so the row is written up
// to that limit (zeros above the diagonal).
__global__ __launch_bounds__(256) void pbuild_kernel(
    const float* __restrict__ logits, const float* __restrict__ colmax,
    const float* __restrict__ colscale, __bf16* __restrict__ ph,
    __bf16* __restrict__ pl) {
  int i = blockIdx.x;
  int b = blockIdx.y;
  int j0 = threadIdx.x * 8;
  int jlimit = (i + 16 + 31) & ~31;  // covers every j the PV K-loop touches
  if (jlimit > TT) jlimit = TT;
  if (j0 >= jlimit) return;

  const float* lrow = logits + (size_t)b * TT * TT + (size_t)i * TT + j0;
  const float* cm = colmax + b * TT + j0;
  const float* cs = colscale + b * TT + j0;
  v8f l = *(const v8f*)lrow;
  v8f mv = *(const v8f*)cm;
  v8f sv = *(const v8f*)cs;

  v8bf vh, vl;
#pragma unroll
  for (int e = 0; e < 8; ++e) {
    int j = j0 + e;
    float p = (j <= i) ? expf(l[e] - mv[e]) * sv[e] : 0.f;
    __bf16 hh = (__bf16)p;
    vh[e] = hh;
    vl[e] = (__bf16)(p - (float)hh);
  }
  size_t idx = (size_t)b * TT * TT + (size_t)i * TT + j0;
  *(v8bf*)(ph + idx) = vh;
  *(v8bf*)(pl + idx) = vl;
}

// --------------------------- kernel 6: PV + residual -----------------------
// out[b,i,v] = x[b,i,v] + sum_{j<=i} P[i,j] * V[j,v].  Pure split-bf16 GEMM:
// A = P rows (hi/lo), B = V^T (hi/lo, contiguous K-strips).  Each wave owns
// 16 i-rows x 128 v-cols (8 tiles, 64 acc VGPRs); K-loop bound i0+16.
__global__ __launch_bounds__(128) void pv_residual_kernel(
    const __bf16* __restrict__ ph, const __bf16* __restrict__ pl,
    const __bf16* __restrict__ vth, const __bf16* __restrict__ vtl,
    const float* __restrict__ x, float* __restrict__ out) {
  int b = blockIdx.z;
  int lane = threadIdx.x & 31;
  int wv = threadIdx.x >> 5;
  int i0 = blockIdx.x * 64 + wv * 16;
  int v0 = blockIdx.y * 128;

  const __bf16* pbh = ph + (size_t)b * TT * TT;
  const __bf16* pbl = pl + (size_t)b * TT * TT;
  const __bf16* vh = vth + (size_t)b * DD * TT;
  const __bf16* vl = vtl + (size_t)b * DD * TT;

  v8f acc[8] = {};
  int kend = i0 + 16;  // need all j <= i0+15 (rest of P row is zero-padded)

  for (int k0 = 0; k0 < kend; k0 += 32) {
    v16bf ah = load_frag_a(pbh, TT, i0, k0, lane);
    v16bf al = load_frag_a(pbl, TT, i0, k0, lane);
#pragma unroll
    for (int t = 0; t < 8; ++t) {
      v16bf bh = load_frag_b(vh, TT, v0 + t * 16, k0, lane);
      v16bf bl = load_frag_b(vl, TT, v0 + t * 16, k0, lane);
      acc[t] = WMMA_BF16(ah, bh, acc[t]);
      acc[t] = WMMA_BF16(ah, bl, acc[t]);
      acc[t] = WMMA_BF16(al, bh, acc[t]);
    }
  }

  int rbase = (lane >> 4) * 8;
#pragma unroll
  for (int t = 0; t < 8; ++t) {
    int c = v0 + t * 16 + (lane & 15);
#pragma unroll
    for (int rr = 0; rr < 8; ++rr) {
      int i = i0 + rbase + rr;
      size_t idx = ((size_t)(b * TT + i)) * DD + c;
      out[idx] = x[idx] + acc[t][rr];
    }
  }
}

// --------------------------- launcher --------------------------------------
extern "C" void kernel_launch(void* const* d_in, const int* in_sizes, int n_in,
                              void* d_out, int out_size, void* d_ws,
                              size_t ws_size, hipStream_t stream) {
  const float* x  = (const float*)d_in[0];
  const float* Wk = (const float*)d_in[1];
  const float* bk = (const float*)d_in[2];
  const float* Wq = (const float*)d_in[3];
  const float* bq = (const float*)d_in[4];
  const float* Wv = (const float*)d_in[5];
  const float* bv = (const float*)d_in[6];
  float* out = (float*)d_out;

  char* ws = (char*)d_ws;
  auto alloc = [&](size_t bytes) -> char* {
    char* p = ws;
    ws += (bytes + 255) & ~(size_t)255;
    return p;
  };

  const size_t n_x = (size_t)NROW * DD;          // 8.4M elems
  const size_t n_w = (size_t)DD * DD;            // 262144 elems
  const size_t n_l = (size_t)BB * TT * TT;       // 33.6M elems

  __bf16* xh  = (__bf16*)alloc(n_x * 2);
  __bf16* xl  = (__bf16*)alloc(n_x * 2);
  __bf16* wkh = (__bf16*)alloc(n_w * 2);
  __bf16* wkl = (__bf16*)alloc(n_w * 2);
  __bf16* wqh = (__bf16*)alloc(n_w * 2);
  __bf16* wql = (__bf16*)alloc(n_w * 2);
  __bf16* wvh = (__bf16*)alloc(n_w * 2);
  __bf16* wvl = (__bf16*)alloc(n_w * 2);
  __bf16* Qh  = (__bf16*)alloc(n_x * 2);
  __bf16* Ql  = (__bf16*)alloc(n_x * 2);
  __bf16* Kh  = (__bf16*)alloc(n_x * 2);
  __bf16* Kl  = (__bf16*)alloc(n_x * 2);
  __bf16* Vth = (__bf16*)alloc(n_x * 2);  // [b][v][t]
  __bf16* Vtl = (__bf16*)alloc(n_x * 2);
  float* colmax   = (float*)alloc((size_t)BB * TT * 4);
  float* colscale = (float*)alloc((size_t)BB * TT * 4);
  float* logits   = (float*)alloc(n_l * 4);     // 134 MB (L2-resident)
  __bf16* Ph      = (__bf16*)alloc(n_l * 2);    // 67 MB
  __bf16* Pl      = (__bf16*)alloc(n_l * 2);    // 67 MB

  // 1) split-convert activations and weights
  cvt_split_kernel<<<(int)((n_x + 255) / 256), 256, 0, stream>>>(x, xh, xl,
                                                                 (int)n_x);
  cvt_split_kernel<<<(int)((n_w + 255) / 256), 256, 0, stream>>>(Wk, wkh, wkl,
                                                                 (int)n_w);
  cvt_split_kernel<<<(int)((n_w + 255) / 256), 256, 0, stream>>>(Wq, wqh, wql,
                                                                 (int)n_w);
  cvt_split_kernel<<<(int)((n_w + 255) / 256), 256, 0, stream>>>(Wv, wvh, wvl,
                                                                 (int)n_w);

  // 2) projections
  dim3 gp(NROW / 64, DD / 64);  // (256, 8)
  proj_wmma_kernel<<<gp, 128, 0, stream>>>(xh, xl, wqh, wql, bq, Qh, Ql, 0);
  proj_wmma_kernel<<<gp, 128, 0, stream>>>(xh, xl, wkh, wkl, bk, Kh, Kl, 0);
  proj_wmma_kernel<<<gp, 128, 0, stream>>>(xh, xl, wvh, wvl, bv, Vth, Vtl, 1);

  // 3) logits (lower-triangular blocks)
  dim3 gl(TT / 64, TT / 64, BB);  // (32, 32, 8)
  logits_wmma_kernel<<<gl, 128, 0, stream>>>(Qh, Ql, Kh, Kl, logits);

  // 4) column softmax stats (axis=1 softmax)
  dim3 gs(TT / 256, BB);
  colstats_kernel<<<gs, 256, 0, stream>>>(logits, colmax, colscale);

  // 5) build P in split bf16 (exp evaluated once per logit)
  dim3 gb(TT, BB);
  pbuild_kernel<<<gb, 256, 0, stream>>>(logits, colmax, colscale, Ph, Pl);

  // 6) PV + residual
  dim3 gv(TT / 64, DD / 128, BB);  // (32, 4, 8)
  pv_residual_kernel<<<gv, 128, 0, stream>>>(Ph, Pl, Vth, Vtl, x, out);
}
